// FusionModel_34041910788395
// MI455X (gfx1250) — matrix operand order repaired and checked by reading, not compiled, not run on hardware
//
#include <hip/hip_runtime.h>

#define A_N   4096
#define EMB   128
#define ORIGD 18
#define KP    32        // ORIG padded to 32 for WMMA K
#define MAXO  16
#define EOBS  32768
#define ECOM  32768
#define NXC   288       // MAX_OBJ * ORIG
#define NSC   17
#define THRES2 (0.02f * 0.02f)

typedef __bf16 bf16;
typedef bf16  v16bf __attribute__((ext_vector_type(16)));
typedef bf16  v8bf  __attribute__((ext_vector_type(8)));
typedef float v8f   __attribute__((ext_vector_type(8)));
typedef unsigned int v4u __attribute__((ext_vector_type(4)));
typedef unsigned int v8u __attribute__((ext_vector_type(8)));

__device__ __forceinline__ bf16 f2bf(float f) {
  unsigned u = __builtin_bit_cast(unsigned, f);
  unsigned r = (u + 0x7FFFu + ((u >> 16) & 1u)) >> 16;   // round-to-nearest-even
  return __builtin_bit_cast(bf16, (unsigned short)r);
}

__device__ __forceinline__ v8f wmma_bf16(v16bf a, v16bf b, v8f c) {
  return __builtin_amdgcn_wmma_f32_16x16x32_bf16(false, a, false, b, (short)0, c,
                                                 false, false);
}

// A fragment 16x32 (MxK), row-major source, 16B-aligned rows.
__device__ __forceinline__ v16bf load_a(const bf16* tile, int rstride, int lane) {
  int r  = lane & 15;
  int kb = (lane & 16) ? 8 : 0;
  const bf16* p = tile + r * rstride + kb;
  v8bf lo = *(const v8bf*)p;
  v8bf hi = *(const v8bf*)(p + 16);
  v16bf a;
#pragma unroll
  for (int e = 0; e < 8; ++e) { a[e] = lo[e]; a[8 + e] = hi[e]; }
  return a;
}

// A fragment with per-row gather (row indices in LDS), rows of width EMB
__device__ __forceinline__ v16bf load_a_gather(const bf16* base, const int* rows,
                                               int k0, int lane) {
  int r  = lane & 15;
  int kb = (lane & 16) ? 8 : 0;
  const bf16* p = base + rows[r] * EMB + k0 + kb;
  v8bf lo = *(const v8bf*)p;
  v8bf hi = *(const v8bf*)(p + 16);
  v16bf a;
#pragma unroll
  for (int e = 0; e < 8; ++e) { a[e] = lo[e]; a[8 + e] = hi[e]; }
  return a;
}

// B fragment from fragment-ordered bf16 weights: [ks][nt][lane][16] contiguous.
// Two 16B vector loads per fragment, zero packing VALU.
__device__ __forceinline__ v16bf load_b_frag(const bf16* wf, int ks, int nt, int NT,
                                             int lane) {
  const bf16* p = wf + ((((ks * NT) + nt) * 32 + lane) << 4);
  v8bf lo = *(const v8bf*)p;
  v8bf hi = *(const v8bf*)(p + 8);
  v16bf b;
#pragma unroll
  for (int e = 0; e < 8; ++e) { b[e] = lo[e]; b[8 + e] = hi[e]; }
  return b;
}

// ---------------------------------------------------------------- kernels ----

__global__ void k_zero(uint4* p, size_t n16) {
  size_t i = (size_t)blockIdx.x * blockDim.x + threadIdx.x;
  uint4 z; z.x = z.y = z.z = z.w = 0u;
  for (; i < n16; i += (size_t)gridDim.x * blockDim.x) p[i] = z;
}

// Convert f32 weight (rowstride N, valid rows kvalid, valid cols ncols) into
// WMMA-B-fragment-ordered bf16. grid=(KS,NT), block=32: lane builds its 16 elems.
__global__ void k_prep(const float* W, int N, int kvalid, int ncols, bf16* dst) {
  int ks = blockIdx.x, nt = blockIdx.y, NT = gridDim.y;
  int lane = threadIdx.x;
  int col = nt * 16 + (lane & 15);
  int kb  = ks * 32 + ((lane & 16) ? 8 : 0);
  v8bf lo, hi;
#pragma unroll
  for (int e = 0; e < 8; ++e) {
    int k0 = kb + e, k1 = kb + 16 + e;
    lo[e] = ((k0 < kvalid) && (col < ncols)) ? f2bf(W[k0 * N + col]) : f2bf(0.f);
    hi[e] = ((k1 < kvalid) && (col < ncols)) ? f2bf(W[k1 * N + col]) : f2bf(0.f);
  }
  bf16* p = dst + ((((ks * NT) + nt) * 32 + lane) << 4);
  *(v8bf*)p = lo;
  *(v8bf*)(p + 8) = hi;
}

// one thread per obs edge: msg = [obj_x[o], obj_pos[o]-agent_pos[a], 0-pad] -> bf16
__global__ void k_build_msg(const float* obj_x, const float* obj_pos,
                            const float* agent_pos, const int* obs_edge, bf16* msg) {
  int e = blockIdx.x * blockDim.x + threadIdx.x;
  if (e >= EOBS) return;
  int a = obs_edge[e];          // row0 = tgt
  int o = obs_edge[EOBS + e];   // row1 = src
  float xv[16];
  const float4* ox = (const float4*)(obj_x + o * 16);
#pragma unroll
  for (int q = 0; q < 4; ++q) {
    float4 t = ox[q];
    xv[q * 4 + 0] = t.x; xv[q * 4 + 1] = t.y;
    xv[q * 4 + 2] = t.z; xv[q * 4 + 3] = t.w;
  }
  v8bf m0, m1, m2, m3;
#pragma unroll
  for (int q = 0; q < 8; ++q) { m0[q] = f2bf(xv[q]); m1[q] = f2bf(xv[8 + q]); }
  float rx = obj_pos[o * 2 + 0] - agent_pos[a * 2 + 0];
  float ry = obj_pos[o * 2 + 1] - agent_pos[a * 2 + 1];
  m2[0] = f2bf(rx); m2[1] = f2bf(ry);
#pragma unroll
  for (int q = 2; q < 8; ++q) m2[q] = f2bf(0.f);
#pragma unroll
  for (int q = 0; q < 8; ++q) m3[q] = f2bf(0.f);
  v8bf* mp = (v8bf*)(msg + e * KP);
  mp[0] = m0; mp[1] = m1; mp[2] = m2; mp[3] = m3;
}

// relu(msg @ W1 + b1) with segment-sum over 8 consecutive edges.
// One wave per 16-edge M-tile = 2 agents; C-fragment row halves ARE the segments.
__global__ void k_enc1(const bf16* msg, const bf16* wf, const float* b1, bf16* hsum) {
  int lane = threadIdx.x;
  int t = blockIdx.x;
  v16bf a = load_a(msg + t * 16 * KP, KP, lane);
  int agent = 2 * t + ((lane & 16) ? 1 : 0);
  int cl = lane & 15;
#pragma unroll
  for (int nt = 0; nt < 8; ++nt) {
    v16bf b = load_b_frag(wf, 0, nt, 8, lane);
    float bias = b1[nt * 16 + cl];
    v8f c;
#pragma unroll
    for (int j = 0; j < 8; ++j) c[j] = bias;
    c = wmma_bf16(a, b, c);
    float s = 0.f;
#pragma unroll
    for (int j = 0; j < 8; ++j) { float v = c[j]; s += (v > 0.f) ? v : 0.f; }
    hsum[agent * EMB + nt * 16 + cl] = f2bf(s);
  }
}

// generic (Mx128) @ (128x128) + bias -> bf16 out. One wave per 16-row tile.
// Fragment-ordered weight (32 KB) staged global->LDS with one TDM DMA.
__global__ void k_gemm128(const bf16* in, const bf16* wf, const float* bias, bf16* out) {
  __shared__ bf16 w[4 * 8 * 32 * 16];        // 32 KB, fragment order
  int lane = threadIdx.x;
  {
    unsigned lds = (unsigned)(size_t)(void*)&w[0];
    unsigned long long ga = (unsigned long long)(size_t)(const void*)wf;
    v4u g0; v8u g1;
    g0[0] = 1u;                                   // count=1, user descriptor
    g0[1] = lds;                                  // lds_addr
    g0[2] = (unsigned)ga;                         // global_addr[31:0]
    g0[3] = (unsigned)(ga >> 32) | (2u << 30);    // global_addr[56:32] | type=2
    const unsigned NEL = 4 * 8 * 32 * 16;         // 16384 bf16 elements
    g1[0] = 1u << 16;                             // data_size=1 (2 bytes/elem)
    g1[1] = (NEL & 0xFFFFu) << 16;                // tensor_dim0[15:0]
    g1[2] = (NEL >> 16) & 0xFFFFu;                // tensor_dim0[31:16], tensor_dim1=0
    g1[3] = (NEL & 0xFFFFu) << 16;                // tile_dim0 = NEL
    g1[4] = 0u;                                   // tile_dim1/2 unused
    g1[5] = NEL;                                  // tensor_dim0_stride[31:0]
    g1[6] = 0u;
    g1[7] = 0u;
    asm volatile("tensor_load_to_lds %0, %1" :: "s"(g0), "s"(g1) : "memory");
    __builtin_amdgcn_s_wait_tensorcnt(0);
  }
  __syncthreads();
  int t = blockIdx.x;
  const bf16* rows = in + t * 16 * EMB;
  int cl = lane & 15, rb = (lane & 16) ? 8 : 0;
  v8f c[8];
#pragma unroll
  for (int nt = 0; nt < 8; ++nt) {
    float bv = bias[nt * 16 + cl];
#pragma unroll
    for (int j = 0; j < 8; ++j) c[nt][j] = bv;
  }
#pragma unroll
  for (int ks = 0; ks < 4; ++ks) {
    v16bf a = load_a(rows + ks * 32, EMB, lane);
#pragma unroll
    for (int nt = 0; nt < 8; ++nt) {
      v16bf b = load_b_frag(w, ks, nt, 8, lane);   // 2x ds_load_b128
      c[nt] = wmma_bf16(a, b, c[nt]);
    }
  }
#pragma unroll
  for (int nt = 0; nt < 8; ++nt)
#pragma unroll
    for (int j = 0; j < 8; ++j)
      out[(t * 16 + rb + j) * EMB + nt * 16 + cl] = f2bf(c[nt][j]);
}

// per comm-edge: scores->npred (argmax), elems = zj@Wx+bx (+relpos), scatter to cand
__global__ void __launch_bounds__(256)
k_mdec(const bf16* enc, const bf16* Wsf, const float* bs, const bf16* Wxf,
       const float* bx, const float* agent_pos, const int* comm_edge,
       bf16* cand, float* posf, unsigned char* valid) {
  __shared__ float scores[8][16][NSC];
  __shared__ float relp[8][16][2];
  __shared__ int   jrow[8][16];
  __shared__ int   npred[8][16];
  int tid = threadIdx.x;
  int w = tid >> 5, lane = tid & 31;
  int m0 = blockIdx.x * 128 + w * 16;
  __builtin_prefetch(Wxf, 0, 1);                  // global_prefetch_b8
  if (lane < 16) {
    int e = m0 + lane;
    int j = comm_edge[e];        // row0 = src
    int i = e >> 3;              // tgt = repeat(arange(A), 8)
    jrow[w][lane] = j;
    relp[w][lane][0] = agent_pos[j * 2 + 0] - agent_pos[i * 2 + 0];
    relp[w][lane][1] = agent_pos[j * 2 + 1] - agent_pos[i * 2 + 1];
  }
  __syncthreads();
  v16bf afr[4];
#pragma unroll
  for (int ks = 0; ks < 4; ++ks) afr[ks] = load_a_gather(enc, jrow[w], ks * 32, lane);
  int cl = lane & 15, rb = (lane & 16) ? 8 : 0;
  // ---- scores (cols 0..16; Wsf zero-padded to 32 cols) ----
  {
    v8f c0, c1;
    float bv0 = bs[cl];
    float bv1 = (16 + cl < NSC) ? bs[16 + cl] : 0.f;
#pragma unroll
    for (int j = 0; j < 8; ++j) { c0[j] = bv0; c1[j] = bv1; }
#pragma unroll
    for (int ks = 0; ks < 4; ++ks) {
      c0 = wmma_bf16(afr[ks], load_b_frag(Wsf, ks, 0, 2, lane), c0);
      c1 = wmma_bf16(afr[ks], load_b_frag(Wsf, ks, 1, 2, lane), c1);
    }
#pragma unroll
    for (int j = 0; j < 8; ++j) {
      scores[w][rb + j][cl] = c0[j];
      if (cl == 0) scores[w][rb + j][16] = c1[j];
    }
  }
  __syncthreads();
  if (lane < 16) {
    float best = scores[w][lane][0]; int bi = 0;
#pragma unroll
    for (int s = 1; s < NSC; ++s) {
      float v = scores[w][lane][s];
      if (v > best) { best = v; bi = s; }
    }
    npred[w][lane] = bi;
  }
  __syncthreads();
  // ---- elems: 18 N-tiles of Wx ----
  for (int nt = 0; nt < 18; ++nt) {
    int n0 = nt * 16;
    v8f c;
    float bv = bx[n0 + cl];
#pragma unroll
    for (int j = 0; j < 8; ++j) c[j] = bv;
#pragma unroll
    for (int ks = 0; ks < 4; ++ks)
      c = wmma_bf16(afr[ks], load_b_frag(Wxf, ks, nt, 18, lane), c);
    int cc = n0 + cl;
    int slot = cc / ORIGD, dim = cc % ORIGD;
#pragma unroll
    for (int j = 0; j < 8; ++j) {
      int rr = rb + j;
      float v = c[j];
      if (dim >= 16) v += relp[w][rr][dim - 16];
      int e = m0 + rr;
      int i = e >> 3, d = e & 7;
      int kc = d * 16 + slot;
      int base = i * 128 + kc;
      cand[base * KP + dim] = f2bf(v);
      if (dim == 16) posf[base * 2 + 0] = v;
      if (dim == 17) posf[base * 2 + 1] = v;
      if (dim == 0)  valid[base] = (slot < npred[w][rr]) ? 1 : 0;
    }
  }
}

// per agent: drop candidates within THRES of an earlier valid one; cap at MAXO-1
__global__ void k_dedup(const float* posf, const unsigned char* valid,
                        unsigned char* keep) {
  __shared__ float px[128], py[128];
  __shared__ int vld[128], scan[128];
  int a = blockIdx.x, k = threadIdx.x;
  int base = a * 128 + k;
  px[k] = posf[base * 2 + 0];
  py[k] = posf[base * 2 + 1];
  vld[k] = valid[base];
  __syncthreads();
  int kp = vld[k];
  if (kp) {
    for (int r = 0; r < k; ++r) {
      if (vld[r]) {
        float dx = px[k] - px[r], dy = py[k] - py[r];
        if (dx * dx + dy * dy < THRES2) { kp = 0; break; }
      }
    }
  }
  scan[k] = kp;
  __syncthreads();
  for (int off = 1; off < 128; off <<= 1) {
    int v = scan[k];
    if (k >= off) v += scan[k - off];
    __syncthreads();
    scan[k] = v;
    __syncthreads();
  }
  if (scan[k] > MAXO - 1) kp = 0;   // inclusive cumsum cap
  keep[base] = (unsigned char)kp;
}

// merged_pre[agent] = sum_kc keep * relu(cand @ menc_W1 + b1). One wave / agent.
__global__ void k_menc1(const bf16* cand, const unsigned char* keep,
                        const bf16* wf, const float* b1, bf16* msum) {
  __shared__ float keepf[16];
  __shared__ float colsum[32];
  int lane = threadIdx.x;
  int agent = blockIdx.x;
  int cl = lane & 15, rb = (lane & 16) ? 8 : 0;
  v16bf bfr[8];
#pragma unroll
  for (int nt = 0; nt < 8; ++nt) bfr[nt] = load_b_frag(wf, 0, nt, 8, lane);
  float sums[8];
#pragma unroll
  for (int nt = 0; nt < 8; ++nt) sums[nt] = 0.f;
  for (int mt = 0; mt < 8; ++mt) {
    __syncthreads();
    if (lane < 16) keepf[lane] = keep[agent * 128 + mt * 16 + lane] ? 1.f : 0.f;
    __syncthreads();
    v16bf a = load_a(cand + (agent * 128 + mt * 16) * KP, KP, lane);
#pragma unroll
    for (int nt = 0; nt < 8; ++nt) {
      float bias = b1[nt * 16 + cl];
      v8f c;
#pragma unroll
      for (int j = 0; j < 8; ++j) c[j] = bias;
      c = wmma_bf16(a, bfr[nt], c);
#pragma unroll
      for (int j = 0; j < 8; ++j) {
        float v = c[j]; v = (v > 0.f) ? v : 0.f;
        sums[nt] += v * keepf[rb + j];
      }
    }
  }
#pragma unroll
  for (int nt = 0; nt < 8; ++nt) {
    __syncthreads();
    colsum[lane] = sums[nt];
    __syncthreads();
    if (lane < 16)
      msum[agent * EMB + nt * 16 + lane] = f2bf(colsum[lane] + colsum[lane + 16]);
  }
}

// final decode: n2 = argmax(merged@Ws+bs); decoded = (merged@Wx+bx) * mask; outputs
__global__ void __launch_bounds__(256)
k_dec(const bf16* merged, const bf16* Wsf, const float* bs, const bf16* Wxf,
      const float* bx, float* out_dec, float* out_batch, float* out_mask) {
  __shared__ float scores[8][16][NSC];
  __shared__ int n2[8][16];
  int tid = threadIdx.x;
  int w = tid >> 5, lane = tid & 31;
  int m0 = blockIdx.x * 128 + w * 16;
  __builtin_prefetch(Wxf, 0, 1);
  const bf16* rows = merged + m0 * EMB;
  v16bf afr[4];
#pragma unroll
  for (int ks = 0; ks < 4; ++ks) afr[ks] = load_a(rows + ks * 32, EMB, lane);
  int cl = lane & 15, rb = (lane & 16) ? 8 : 0;
  {
    v8f c0, c1;
    float bv0 = bs[cl];
    float bv1 = (16 + cl < NSC) ? bs[16 + cl] : 0.f;
#pragma unroll
    for (int j = 0; j < 8; ++j) { c0[j] = bv0; c1[j] = bv1; }
#pragma unroll
    for (int ks = 0; ks < 4; ++ks) {
      c0 = wmma_bf16(afr[ks], load_b_frag(Wsf, ks, 0, 2, lane), c0);
      c1 = wmma_bf16(afr[ks], load_b_frag(Wsf, ks, 1, 2, lane), c1);
    }
#pragma unroll
    for (int j = 0; j < 8; ++j) {
      scores[w][rb + j][cl] = c0[j];
      if (cl == 0) scores[w][rb + j][16] = c1[j];
    }
  }
  __syncthreads();
  if (lane < 16) {
    float best = scores[w][lane][0]; int bi = 0;
#pragma unroll
    for (int s = 1; s < NSC; ++s) {
      float v = scores[w][lane][s];
      if (v > best) { best = v; bi = s; }
    }
    n2[w][lane] = bi;
  }
  __syncthreads();
  for (int nt = 0; nt < 18; ++nt) {
    int n0 = nt * 16;
    v8f c;
    float bv = bx[n0 + cl];
#pragma unroll
    for (int j = 0; j < 8; ++j) c[j] = bv;
#pragma unroll
    for (int ks = 0; ks < 4; ++ks)
      c = wmma_bf16(afr[ks], load_b_frag(Wxf, ks, nt, 18, lane), c);
    int cc = n0 + cl;
    int slot = cc / ORIGD, dim = cc % ORIGD;
#pragma unroll
    for (int j = 0; j < 8; ++j) {
      int rr = rb + j;
      int i = m0 + rr;
      float v = (slot < n2[w][rr]) ? c[j] : 0.f;
      out_dec[(i * MAXO + slot) * ORIGD + dim] = v;
    }
  }
  if (lane < 16) {
    int i = m0 + lane, nn = n2[w][lane];
#pragma unroll
    for (int s = 0; s < MAXO; ++s) {
      out_batch[i * MAXO + s] = (float)i;
      out_mask[i * MAXO + s] = (s < nn) ? 1.f : 0.f;
    }
  }
}

// -------------------------------------------------------------- launcher ----

extern "C" void kernel_launch(void* const* d_in, const int* in_sizes, int n_in,
                              void* d_out, int out_size, void* d_ws, size_t ws_size,
                              hipStream_t stream) {
  const float* obj_x     = (const float*)d_in[0];
  const float* obj_pos   = (const float*)d_in[1];
  const float* agent_pos = (const float*)d_in[2];
  const float* enc_W1    = (const float*)d_in[3];
  const float* enc_b1    = (const float*)d_in[4];
  const float* enc_W2    = (const float*)d_in[5];
  const float* enc_b2    = (const float*)d_in[6];
  const float* mdec_Ws   = (const float*)d_in[7];
  const float* mdec_bs   = (const float*)d_in[8];
  const float* mdec_Wx   = (const float*)d_in[9];
  const float* mdec_bx   = (const float*)d_in[10];
  const float* menc_W1   = (const float*)d_in[11];
  const float* menc_b1   = (const float*)d_in[12];
  const float* menc_W2   = (const float*)d_in[13];
  const float* menc_b2   = (const float*)d_in[14];
  const float* dec_Ws    = (const float*)d_in[15];
  const float* dec_bs    = (const float*)d_in[16];
  const float* dec_Wx    = (const float*)d_in[17];
  const float* dec_bx    = (const float*)d_in[18];
  const int*   obs_edge  = (const int*)d_in[19];
  const int*   comm_edge = (const int*)d_in[20];

  char* ws = (char*)d_ws;
  bf16* msg    = (bf16*)(ws + 0);            //  2 MB : 32768 x 32 bf16
  bf16* hsum   = (bf16*)(ws + 2097152);      //  1 MB : 4096 x 128 bf16
  bf16* enc    = (bf16*)(ws + 3145728);      //  1 MB
  bf16* cand   = (bf16*)(ws + 4194304);      // 32 MB : 4096 x 128 x 32 bf16
  float* posf  = (float*)(ws + 37748736);    //  4 MB
  unsigned char* valid = (unsigned char*)(ws + 41943040);  // 512 KB
  unsigned char* keep  = (unsigned char*)(ws + 42467328);  // 512 KB
  bf16* msum   = (bf16*)(ws + 42991616);     //  1 MB
  bf16* merged = (bf16*)(ws + 44040192);     //  1 MB
  // fragment-ordered bf16 weights
  bf16* encW1f  = (bf16*)(ws + 45088768);    //  8 KB : 1ks x 8nt
  bf16* encW2f  = (bf16*)(ws + 45096960);    // 32 KB : 4ks x 8nt
  bf16* mdecWsf = (bf16*)(ws + 45129728);    //  8 KB : 4ks x 2nt
  bf16* mdecWxf = (bf16*)(ws + 45137920);    // 72 KB : 4ks x 18nt
  bf16* mencW1f = (bf16*)(ws + 45211648);    //  8 KB
  bf16* mencW2f = (bf16*)(ws + 45219840);    // 32 KB
  bf16* decWsf  = (bf16*)(ws + 45252608);    //  8 KB
  bf16* decWxf  = (bf16*)(ws + 45260800);    // 72 KB

  float* out       = (float*)d_out;
  float* out_dec   = out;                          // 4096*16*18
  float* out_batch = out + A_N * MAXO * ORIGD;     // 4096*16
  float* out_mask  = out_batch + A_N * MAXO;       // 4096*16

  // weight conversion to WMMA fragment order (bf16, zero-padded)
  k_prep<<<dim3(1, 8),  32, 0, stream>>>(enc_W1, 128, ORIGD, 128, encW1f);
  k_prep<<<dim3(4, 8),  32, 0, stream>>>(enc_W2, 128, 128, 128, encW2f);
  k_prep<<<dim3(4, 2),  32, 0, stream>>>(mdec_Ws, NSC, 128, NSC, mdecWsf);
  k_prep<<<dim3(4, 18), 32, 0, stream>>>(mdec_Wx, NXC, 128, NXC, mdecWxf);
  k_prep<<<dim3(1, 8),  32, 0, stream>>>(menc_W1, 128, ORIGD, 128, mencW1f);
  k_prep<<<dim3(4, 8),  32, 0, stream>>>(menc_W2, 128, 128, 128, mencW2f);
  k_prep<<<dim3(4, 2),  32, 0, stream>>>(dec_Ws, NSC, 128, NSC, decWsf);
  k_prep<<<dim3(4, 18), 32, 0, stream>>>(dec_Wx, NXC, 128, NXC, decWxf);

  k_zero<<<4096, 256, 0, stream>>>((uint4*)cand, (size_t)(A_N * 128 * KP * 2 / 16));
  k_build_msg<<<EOBS / 256, 256, 0, stream>>>(obj_x, obj_pos, agent_pos, obs_edge, msg);
  k_enc1<<<EOBS / 16, 32, 0, stream>>>(msg, encW1f, enc_b1, hsum);
  k_gemm128<<<A_N / 16, 32, 0, stream>>>(hsum, encW2f, enc_b2, enc);
  k_mdec<<<ECOM / 128, 256, 0, stream>>>(enc, mdecWsf, mdec_bs, mdecWxf, mdec_bx,
                                         agent_pos, comm_edge, cand, posf, valid);
  k_dedup<<<A_N, 128, 0, stream>>>(posf, valid, keep);
  k_menc1<<<A_N, 32, 0, stream>>>(cand, keep, mencW1f, menc_b1, msum);
  k_gemm128<<<A_N / 16, 32, 0, stream>>>(msum, mencW2f, menc_b2, merged);
  k_dec<<<A_N / 128, 256, 0, stream>>>(merged, decWsf, dec_bs, decWxf, dec_bx,
                                       out_dec, out_batch, out_mask);
}
